// LearnableUpsamplingLayer_44229573214792
// MI455X (gfx1250) — matrix operand order repaired and checked by reading, not compiled, not run on hardware
//
#include <hip/hip_runtime.h>
#include <hip/hip_bf16.h>
#include <math.h>

typedef _Float16 v16h __attribute__((ext_vector_type(16)));
typedef _Float16 v8h  __attribute__((ext_vector_type(8)));
typedef _Float16 v4h  __attribute__((ext_vector_type(4)));
typedef float    v8f  __attribute__((ext_vector_type(8)));
typedef float    v4f  __attribute__((ext_vector_type(4)));

#define B_    8
#define T_    16384
#define C_    64
#define NF    64
#define FS    5
#define TWOT  (2*T_)          // 32768
#define KDIM  (FS*C_)         // 320
#define KSTEPS (KDIM/32)      // 10
#define MTILE 128             // output rows per workgroup
#define UROWS (MTILE+4)       // 132 staged up-rows (halo of 2 each side)
#define SLOPE 0.3f

// d_ws layout: [0, 40960) : f16 packed B fragments (K=320 x N=64)
//              [40960, 41216) : float sigmoid(interp_weights)[64]
#define WSIG_OFF (KDIM*NF)    // in _Float16 units *2 bytes = 40960

// ---------------------------------------------------------------------------
// Pre-kernel: pack conv kernel into per-lane WMMA B-fragment layout (f16) and
// compute per-channel sigmoid gate.
// B fragment (s,nt): lane L holds 16 halves e=0..15 with
//   K = s*32 + 16*(L/16) + e,  N = nt*16 + (L%16)
// stored contiguously: idx = ((s*4+nt)*32 + L)*16 + e
// ---------------------------------------------------------------------------
__global__ void prep_kernel(const float* __restrict__ iw,
                            const float* __restrict__ kern,
                            _Float16* __restrict__ bpack,
                            float* __restrict__ wsig) {
    int idx = blockIdx.x * 256 + threadIdx.x;
    if (idx < KDIM * NF) {
        int e    = idx & 15;
        int lane = (idx >> 4) & 31;
        int nt   = (idx >> 9) & 3;
        int s    = idx >> 11;
        int h    = lane >> 4;
        int n16  = lane & 15;
        int kg   = s * 32 + 16 * h + e;   // global K (= tap*64 + c)
        int n    = nt * 16 + n16;
        bpack[idx] = (_Float16)kern[kg * NF + n];  // kernel is [5,64,64] = W[320,64]
    } else if (idx < KDIM * NF + C_) {
        int c = idx - KDIM * NF;
        float v = iw[c];
        wsig[c] = 1.0f / (1.0f + expf(-v));
    }
}

// ---------------------------------------------------------------------------
// Main kernel: upsample into LDS (f16) + WMMA GEMM conv + bias + leaky relu
// ---------------------------------------------------------------------------
__global__ void __launch_bounds__(128) upconv_kernel(
        const float* __restrict__ x,        // [8,16384,64]
        const float* __restrict__ bias,     // [64]
        const _Float16* __restrict__ bpack, // packed weights in d_ws
        const float* __restrict__ wsig,     // sigmoid gate in d_ws
        float* __restrict__ out) {          // [8,32768,64]
    __shared__ _Float16 Bs[KDIM * NF];      // 40960 B
    __shared__ _Float16 Us[UROWS * C_];     // 16896 B

    const int tid  = threadIdx.x;
    const int b    = blockIdx.x >> 8;       // 256 tiles per batch
    const int tile = blockIdx.x & 255;
    const int p0   = tile * MTILE;          // first output row of this tile

    // ---- stage packed B weights into LDS (2560 x 16B units) ----
    {
        const v4f* src = (const v4f*)bpack;
        v4f*       dst = (v4f*)Bs;
        #pragma unroll
        for (int j = 0; j < 20; ++j)
            dst[j * 128 + tid] = src[j * 128 + tid];
    }

    // ---- build f16 up-tile in LDS: rows p0-2 .. p0+129 (SAME halo) ----
    {
        const int items = UROWS * (C_ / 4);   // 2112 quad-channel items
        const v4f one = {1.f, 1.f, 1.f, 1.f};
        for (int q = tid; q < items; q += 128) {
            int r  = q >> 4;
            int c4 = (q & 15) << 2;
            int p  = p0 - 2 + r;
            v4f up = {0.f, 0.f, 0.f, 0.f};
            if (p >= 0 && p < TWOT) {
                int i = p >> 1;
                const v4f* xrow = (const v4f*)(x + ((size_t)b * T_ + i) * C_);
                v4f xi = xrow[c4 >> 2];
                if (p & 1) {
                    v4f xn = {0.f, 0.f, 0.f, 0.f};
                    if (i + 1 < T_)
                        xn = ((const v4f*)(x + ((size_t)b * T_ + i + 1) * C_))[c4 >> 2];
                    v4f w = *(const v4f*)(wsig + c4);
                    up = w * xi + (one - w) * xn;
                } else {
                    up = xi;
                }
            }
            *(v4h*)(Us + r * C_ + c4) = __builtin_convertvector(up, v4h);
        }
    }
    __syncthreads();

    const int wave = tid >> 5;
    const int lane = tid & 31;
    const int h    = lane >> 4;   // half-wave group
    const int m    = lane & 15;   // row within A tile / col within D tile
    const int r0   = wave * 32;   // this wave's first row within the tile

    v8f acc[2][4];
    #pragma unroll
    for (int mt = 0; mt < 2; ++mt)
        #pragma unroll
        for (int nt = 0; nt < 4; ++nt)
            acc[mt][nt] = (v8f){0.f,0.f,0.f,0.f,0.f,0.f,0.f,0.f};

    #pragma unroll
    for (int s = 0; s < KSTEPS; ++s) {
        // Each lane's 16 A halves = two contiguous 8-channel runs:
        //   run0: K = s*32 + 16h + {0..7}  -> VGPRs 0..3 (elements 0..7)
        //   run1: K = s*32 + 16h + 8 + {0..7} -> VGPRs 4..7 (elements 8..15)
        int kg0 = s * 32 + 16 * h;
        int kg1 = kg0 + 8;
        int tap0 = kg0 >> 6, c0 = kg0 & 63;
        int tap1 = kg1 >> 6, c1 = kg1 & 63;

        v16h a[2];
        #pragma unroll
        for (int mt = 0; mt < 2; ++mt) {
            int ub = r0 + mt * 16 + m;   // A row m maps to up-row (p + tap - 2) = Us[ub + tap]
            v8h lo = *(const v8h*)(Us + (ub + tap0) * C_ + c0);
            v8h hi = *(const v8h*)(Us + (ub + tap1) * C_ + c1);
            a[mt] = __builtin_shufflevector(lo, hi,
                     0,1,2,3,4,5,6,7,8,9,10,11,12,13,14,15);
        }
        #pragma unroll
        for (int nt = 0; nt < 4; ++nt) {
            v16h bf = *(const v16h*)(Bs + ((size_t)(s * 4 + nt) * 32 + lane) * 16);
            acc[0][nt] = __builtin_amdgcn_wmma_f32_16x16x32_f16(
                false, a[0], false, bf, (short)0, acc[0][nt], false, false);
            acc[1][nt] = __builtin_amdgcn_wmma_f32_16x16x32_f16(
                false, a[1], false, bf, (short)0, acc[1][nt], false, false);
        }
    }

    // ---- epilogue: bias + leaky relu, write out ----
    // D layout: lane holds N = nt*16 + m, VGPR v holds row M = v + 8*h
    #pragma unroll
    for (int nt = 0; nt < 4; ++nt) {
        int n = nt * 16 + m;
        float bv = bias[n];
        #pragma unroll
        for (int mt = 0; mt < 2; ++mt) {
            size_t rowbase = (size_t)b * TWOT + p0 + r0 + mt * 16 + 8 * h;
            #pragma unroll
            for (int v = 0; v < 8; ++v) {
                float y = acc[mt][nt][v] + bv;
                y = (y >= 0.f) ? y : SLOPE * y;
                out[(rowbase + v) * NF + n] = y;
            }
        }
    }
}

// ---------------------------------------------------------------------------
extern "C" void kernel_launch(void* const* d_in, const int* in_sizes, int n_in,
                              void* d_out, int out_size, void* d_ws, size_t ws_size,
                              hipStream_t stream) {
    const float* x    = (const float*)d_in[0];  // [8,16384,64]
    const float* iw   = (const float*)d_in[1];  // [64]
    const float* kern = (const float*)d_in[2];  // [5,64,64]
    const float* bias = (const float*)d_in[3];  // [64]
    float*       out  = (float*)d_out;

    _Float16* bpack = (_Float16*)d_ws;
    float*    wsig  = (float*)((char*)d_ws + (size_t)KDIM * NF * sizeof(_Float16));

    // 20480 packed weight elems + 64 sigmoid values, 256 threads/block
    prep_kernel<<<84, 256, 0, stream>>>(iw, kern, bpack, wsig);

    // 8 batches * 256 tiles of 128 rows, 128 threads (4 waves) each
    upconv_kernel<<<B_ * (TWOT / MTILE), 128, 0, stream>>>(x, bias, bpack, wsig, out);
}
